// QLSTM_65481071404254
// MI455X (gfx1250) — compile-verified, matrix operand order checked by reading, not run on hardware
//
#include <hip/hip_runtime.h>

// ---------------- problem constants ----------------
#define T_STEPS 2048
#define BATCH   32
#define DIN     512
#define DHID    512
#define KDIM    1024      // DIN + DHID
#define NBLK    64        // persistent blocks; each owns 8 hidden columns
#define QCOLS   8         // hidden cols per block
#define NLOC    32        // local gate cols = 4*QCOLS  (f|i|g|o)
#define NTHREADS 256      // 8 wave32 waves
#define KTILES  32        // KDIM / 32
#define A_PAD   1032      // row pitch (elements) for combined operand (bank-conflict pad)

typedef __attribute__((ext_vector_type(16))) __bf16 v16bf;
typedef __attribute__((ext_vector_type(8)))  __bf16 v8bf;
typedef __attribute__((ext_vector_type(4)))  __bf16 v4bf;
typedef __attribute__((ext_vector_type(8)))  float  v8f;

// ~146 KB dynamic LDS (fits the 320 KB CDNA5 WGP LDS, one block per WGP)
struct __align__(32) Smem {
  __bf16 W[KTILES][2][32][16];   // 64 KB: B operand, pre-swizzled fragment-major (32B/lane, contiguous)
  __bf16 A[BATCH][A_PAD];        // 66 KB: [x_t | h_{t-1}] bf16, padded rows
  float  Gp[4][BATCH][NLOC];     // 16 KB: per-K-split partial gates (f32 accum)
  float  C[BATCH][QCOLS];        // cell state (persistent across steps)
  float  Bq[NLOC];               // bias slice
};

__global__ void qlstm_init(unsigned* bar, __bf16* hbuf) {
  int i = blockIdx.x * blockDim.x + threadIdx.x;
  if (i == 0) *bar = 0u;
  if (i < 2 * BATCH * DHID) hbuf[i] = (__bf16)0.f;   // h0 = 0, both ping-pong buffers
}

__global__ __launch_bounds__(NTHREADS)
void qlstm_persistent(const float* __restrict__ x,
                      const float* __restrict__ Wf, const float* __restrict__ bfv,
                      const float* __restrict__ Wi, const float* __restrict__ biv,
                      const float* __restrict__ Wg, const float* __restrict__ bgv,
                      const float* __restrict__ Wo, const float* __restrict__ bov,
                      float* __restrict__ out,
                      unsigned* __restrict__ bar,
                      __bf16* __restrict__ hbuf)
{
  extern __shared__ char smem_raw[];
  Smem& S = *reinterpret_cast<Smem*>(smem_raw);

  const int tid = threadIdx.x;
  const int blk = blockIdx.x;

  // ---- one-time: swizzle this block's W4 column slice into LDS, bf16, B-fragment order
  // B frag (16-bit, 32x16 KxN): lane<16 -> N=lane,   K = kb..kb+7 (e0..7), kb+16..kb+23 (e8..15), kb=32*kt
  //                             lane>=16 -> N=lane-16, same with kb+8
  const float* Wp[4] = {Wf, Wi, Wg, Wo};
  for (int idx = tid; idx < KTILES * 2 * 32 * 16; idx += NTHREADS) {
    int e    = idx & 15;
    int lane = (idx >> 4) & 31;
    int nt   = (idx >> 9) & 1;
    int kt   = idx >> 10;
    int nloc = nt * 16 + (lane & 15);
    int gate = nloc >> 3, q = nloc & 7;
    int kb   = (kt << 5) + ((lane >> 4) & 1) * 8;
    int k    = kb + (e & 7) + ((e >> 3) << 4);
    S.W[kt][nt][lane][e] = (__bf16)Wp[gate][k * DHID + blk * QCOLS + q];
  }
  const float* bp[4] = {bfv, biv, bgv, bov};
  if (tid < NLOC) S.Bq[tid] = bp[tid >> 3][blk * QCOLS + (tid & 7)];
  { int m = tid >> 3, q = tid & 7; S.C[m][q] = 0.f; }
  __syncthreads();

  // wave mapping: 8 waves = 2 M-tiles x 4 K-splits; each wave does both N-tiles
  const int wv   = tid >> 5;
  const int lane = tid & 31;
  const int mtl  = wv & 1;            // M tile (batch rows 0-15 / 16-31)
  const int kh   = wv >> 1;           // K split 0..3 (8 k-tiles each)
  const int rowA = mtl * 16 + (lane & 15);
  const int alo  = ((lane >> 4) & 1) * 8;            // A-frag lane-half K offset
  const int r0   = mtl * 16 + ((lane >> 4) & 1) * 8; // C-frag row base
  const int cn   = lane & 15;
  const int m_e  = tid >> 3, q_e = tid & 7;          // elementwise mapping (m, q)
  float h_last = 0.f;

  for (int t = 0; t < T_STEPS; ++t) {
    // ---- stage combined operand A = [x_t | h_{t-1}] as bf16 in LDS
    const float* xt = x + (size_t)t * (BATCH * DIN);
    for (int i = tid; i < BATCH * DIN / 4; i += NTHREADS) {
      const float4 v = ((const float4*)xt)[i];
      int el = i << 2;
      int m = el >> 9, k = el & (DIN - 1);
      v4bf b; b[0] = (__bf16)v.x; b[1] = (__bf16)v.y; b[2] = (__bf16)v.z; b[3] = (__bf16)v.w;
      *(v4bf*)&S.A[m][k] = b;
    }
    const __bf16* hr = hbuf + (size_t)(t & 1) * (BATCH * DHID);
    for (int i = tid; i < BATCH * DHID / 8; i += NTHREADS) {
      int el = i << 3;
      int m = el >> 9, k = el & (DHID - 1);
      *(v8bf*)&S.A[m][DIN + k] = *(const v8bf*)(hr + el);
    }
    if (t + 1 < T_STEPS)
      __builtin_prefetch(xt + BATCH * DIN + tid * 64, 0, 1);  // global_prefetch_b8
    __syncthreads();

    // ---- WMMA: partial gates for this K split
    v8f acc0 = {0.f,0.f,0.f,0.f,0.f,0.f,0.f,0.f};
    v8f acc1 = {0.f,0.f,0.f,0.f,0.f,0.f,0.f,0.f};
    #pragma unroll
    for (int i = 0; i < 8; ++i) {
      int kt = kh * 8 + i;
      int k0 = (kt << 5) + alo;
      // A fragment: two 16B chunks (K k0..k0+7 and k0+16..k0+23) per lane
      v8bf al = *(const v8bf*)&S.A[rowA][k0];
      v8bf ah = *(const v8bf*)&S.A[rowA][k0 + 16];
      v16bf a = __builtin_shufflevector(al, ah, 0,1,2,3,4,5,6,7,8,9,10,11,12,13,14,15);
      // B fragments: 32 contiguous bytes per lane -> single v16bf (2x ds_load_b128, folded offsets)
      v16bf b0 = *(const v16bf*)&S.W[kt][0][lane][0];
      v16bf b1 = *(const v16bf*)&S.W[kt][1][lane][0];
      acc0 = __builtin_amdgcn_wmma_f32_16x16x32_bf16(false, a, false, b0, (short)0, acc0, false, false);
      acc1 = __builtin_amdgcn_wmma_f32_16x16x32_bf16(false, a, false, b1, (short)0, acc1, false, false);
    }
    // C frag layout: VGPR r -> (M = r0+r, N = cn) for each lane half
    #pragma unroll
    for (int r = 0; r < 8; ++r) {
      S.Gp[kh][r0 + r][cn]      = acc0[r];
      S.Gp[kh][r0 + r][16 + cn] = acc1[r];
    }
    __syncthreads();

    // ---- elementwise LSTM cell (one thread per (batch m, hidden q))
    {
      float fg = S.Gp[0][m_e][q_e]      + S.Gp[1][m_e][q_e]      + S.Gp[2][m_e][q_e]      + S.Gp[3][m_e][q_e]      + S.Bq[q_e];
      float ig = S.Gp[0][m_e][8 + q_e]  + S.Gp[1][m_e][8 + q_e]  + S.Gp[2][m_e][8 + q_e]  + S.Gp[3][m_e][8 + q_e]  + S.Bq[8 + q_e];
      float gg = S.Gp[0][m_e][16 + q_e] + S.Gp[1][m_e][16 + q_e] + S.Gp[2][m_e][16 + q_e] + S.Gp[3][m_e][16 + q_e] + S.Bq[16 + q_e];
      float og = S.Gp[0][m_e][24 + q_e] + S.Gp[1][m_e][24 + q_e] + S.Gp[2][m_e][24 + q_e] + S.Gp[3][m_e][24 + q_e] + S.Bq[24 + q_e];
      float sf = 1.f / (1.f + __expf(-fg));
      float si = 1.f / (1.f + __expf(-ig));
      float so = 1.f / (1.f + __expf(-og));
      float c  = sf * S.C[m_e][q_e] + si * tanhf(gg);
      S.C[m_e][q_e] = c;
      float h  = so * tanhf(c);
      h_last = h;
      __bf16* hw = hbuf + (size_t)((t + 1) & 1) * (BATCH * DHID);
      hw[m_e * DHID + blk * QCOLS + q_e] = (__bf16)h;                         // broadcast for next step
      out[((size_t)t * BATCH + m_e) * DHID + blk * QCOLS + q_e] = h;          // stacked output
    }

    // ---- grid-wide barrier: publish h_t, then proceed to step t+1
    __threadfence();
    __syncthreads();
    if (tid == 0) {
      __hip_atomic_fetch_add(bar, 1u, __ATOMIC_RELEASE, __HIP_MEMORY_SCOPE_AGENT);
      unsigned want = (unsigned)(t + 1) * (unsigned)gridDim.x;
      while (__hip_atomic_load(bar, __ATOMIC_ACQUIRE, __HIP_MEMORY_SCOPE_AGENT) < want)
        __builtin_amdgcn_s_sleep(1);
    }
    __syncthreads();
  }

  // ---- final (hx, cx) appended after stacked outputs
  {
    size_t base = (size_t)T_STEPS * BATCH * DHID;
    out[base + m_e * DHID + blk * QCOLS + q_e] = h_last;
    out[base + (size_t)BATCH * DHID + m_e * DHID + blk * QCOLS + q_e] = S.C[m_e][q_e];
  }
}

extern "C" void kernel_launch(void* const* d_in, const int* in_sizes, int n_in,
                              void* d_out, int out_size, void* d_ws, size_t ws_size,
                              hipStream_t stream) {
  (void)in_sizes; (void)n_in; (void)out_size; (void)ws_size;
  const float* x   = (const float*)d_in[0];
  const float* Wf  = (const float*)d_in[1];
  const float* bfv = (const float*)d_in[2];
  const float* Wi  = (const float*)d_in[3];
  const float* biv = (const float*)d_in[4];
  const float* Wg  = (const float*)d_in[5];
  const float* bgv = (const float*)d_in[6];
  const float* Wo  = (const float*)d_in[7];
  const float* bov = (const float*)d_in[8];
  float* out = (float*)d_out;

  unsigned* bar = (unsigned*)d_ws;                         // barrier counter @ ws+0
  __bf16*   hbuf = (__bf16*)((char*)d_ws + 256);           // ping-pong h: 2 x 32x512 bf16

  qlstm_init<<<(2 * BATCH * DHID + 255) / 256, 256, 0, stream>>>(bar, hbuf);
  qlstm_persistent<<<NBLK, NTHREADS, sizeof(Smem), stream>>>(
      x, Wf, bfv, Wi, biv, Wg, bgv, Wo, bov, out, bar, hbuf);
}
